// ICENodeUniform_59124519797163
// MI455X (gfx1250) — compile-verified, matrix operand order checked by reading, not run on hardware
//
#include <hip/hip_runtime.h>
#include <hip/hip_bf16.h>

typedef __bf16 bf16;
typedef __attribute__((ext_vector_type(16))) __bf16 v16bf;
typedef __attribute__((ext_vector_type(8)))  float  v8f;

#define Bsz   512
#define Tlen  24
#define Vocab 8000
#define Edim  300
#define Mdim  30
#define Hdim  330
#define HP    352          // padded H: 11 K-chunks of 32, 22 N-tiles of 16
#define EP    320          // padded E: 10 K-chunks of 32, 20 N-tiles of 16
#define NSTEP 8
__device__ __constant__ float DTc = 7.0f / 8.0f;

// Scalar (SGPR) wave index so tile loops become s_cbranch, keeping EXEC all-1s
// around every WMMA (ISA 7.12 requires EXEC==all ones for WMMA).
__device__ __forceinline__ int wave_id() {
    return __builtin_amdgcn_readfirstlane((int)(threadIdx.x >> 5));
}

// --- element->K mapping for 16-bit WMMA A/B fragments (ISA 7.12.2) --------
__device__ __forceinline__ int kmap_elem(int e, int half) {
    int v = e >> 1, p = e & 1;
    int base = (v < 4) ? (2 * v) : (16 + 2 * (v - 4));
    return base + 8 * half + p;
}

// Load A fragment (16 rows x 32 K, bf16) from LDS row-major [16][ld]
__device__ __forceinline__ v16bf load_a_frag(const bf16* s, int ld, int k0) {
    int lane = threadIdx.x & 31;
    int row = lane & 15, half = (lane >> 4) & 1;
    v16bf a;
#pragma unroll
    for (int e = 0; e < 16; ++e)
        a[e] = s[row * ld + k0 + kmap_elem(e, half)];
    return a;
}

// Load B fragment from fragment-major packed weights (32 bytes / lane)
__device__ __forceinline__ v16bf load_b_frag(const bf16* p) {
    int lane = threadIdx.x & 31;
    return *(const v16bf*)(p + lane * 16);
}

__device__ __forceinline__ v8f wmma_bf16(v16bf a, v16bf b, v8f c) {
    return __builtin_amdgcn_wmma_f32_16x16x32_bf16(false, a, false, b, (short)0, c,
                                                   false, false);
}

// Tiled GEMM over one 16-row LDS tile: dst = act(src @ Wpacked)
// mode 0: tanh -> dstB(bf16); mode 1: tanh -> dstF(f32); mode 2: raw -> dstF
__device__ __forceinline__ void gemm_tile(const bf16* srcB, int ld, const bf16* Wp,
                                          int ktiles, int ntiles, bf16* dstB,
                                          float* dstF, int ldf, int mode) {
    int wave = wave_id();                         // SGPR -> scalar branches
    int lane = threadIdx.x & 31;
    for (int nt = wave; nt < ntiles; nt += 8) {
        v8f acc = {};
        for (int kc = 0; kc < ktiles; ++kc) {
            v16bf a = load_a_frag(srcB, ld, kc * 32);
            v16bf b = load_b_frag(Wp + (((size_t)(kc * ntiles + nt)) << 9));
            acc = wmma_bf16(a, b, acc);
        }
        int col = nt * 16 + (lane & 15);
#pragma unroll
        for (int r = 0; r < 8; ++r) {
            int m = r + 8 * (lane >> 4);
            float v = acc[r];
            if (mode != 2) v = tanhf(v);
            if (dstB) dstB[m * ld + col] = (bf16)v;
            if (dstF) dstF[m * ldf + col] = v;
        }
    }
}

// --- pack fp32 weight (Nreal x Kreal, row-major) into fragment-major bf16 --
__global__ __launch_bounds__(256) void pack_b(const float* __restrict__ W,
                                              bf16* __restrict__ dst, int Kreal,
                                              int Nreal, int ktiles, int ntiles) {
    int total = ktiles * ntiles * 512;
    for (int g = blockIdx.x * 256 + threadIdx.x; g < total; g += gridDim.x * 256) {
        int e = g & 15, lane = (g >> 4) & 31, tile = g >> 9;
        int nt = tile % ntiles, kt = tile / ntiles;
        int k = kt * 32 + kmap_elem(e, lane >> 4);
        int n = nt * 16 + (lane & 15);
        float v = (k < Kreal && n < Nreal) ? W[(size_t)n * Kreal + k] : 0.0f;
        dst[g] = (bf16)v;
    }
}

// --- g_all = codes @ W_emb^T + b_emb  (rows = B*T, padded cols = EP) -------
// 20 N-tiles over 8 waves: every wave owns tiles {wave, wave+8}; waves 0..3
// additionally own wave+16. Named accumulators -> no VGPR-array indexing.
__global__ __launch_bounds__(256) void embed_kernel(const float* __restrict__ codes,
                                                    const bf16* __restrict__ Wembp,
                                                    const float* __restrict__ b_emb,
                                                    float* __restrict__ g_all) {
    __shared__ bf16 As[16][32];
    int row0 = blockIdx.x * 16;
    int wave = wave_id();                         // SGPR
    int lane = threadIdx.x & 31;
    v8f acc0 = {}, acc1 = {}, acc2 = {};
    const int NT = 20, KT = Vocab / 32;           // 250
    bool has2 = (wave < 4);                       // scalar condition
    for (int kc = 0; kc < KT; ++kc) {
        __syncthreads();
        for (int i = threadIdx.x; i < 512; i += 256) {
            int r = i >> 5, c = i & 31;
            As[r][c] = (bf16)codes[(size_t)(row0 + r) * Vocab + kc * 32 + c];
        }
        __syncthreads();
        v16bf a = load_a_frag(&As[0][0], 32, 0);
        const bf16* wrow = Wembp + (((size_t)kc * NT) << 9);
        v16bf b0 = load_b_frag(wrow + ((size_t)(wave) << 9));
        acc0 = wmma_bf16(a, b0, acc0);
        v16bf b1 = load_b_frag(wrow + ((size_t)(wave + 8) << 9));
        acc1 = wmma_bf16(a, b1, acc1);
        if (has2) {                               // s_cbranch, EXEC untouched
            v16bf b2 = load_b_frag(wrow + ((size_t)(wave + 16) << 9));
            acc2 = wmma_bf16(a, b2, acc2);
        }
    }
    int colbase = (lane & 15);
#pragma unroll
    for (int t = 0; t < 3; ++t) {
        if (t == 2 && !has2) break;               // scalar
        int nt = wave + 8 * t;
        int col = nt * 16 + colbase;
        float bias = (col < Edim) ? b_emb[col] : 0.0f;
        v8f acc = (t == 0) ? acc0 : ((t == 1) ? acc1 : acc2);
#pragma unroll
        for (int r = 0; r < 8; ++r) {
            int m = r + 8 * (lane >> 4);
            float v = (col < Edim) ? (acc[r] + bias) : 0.0f;
            g_all[(size_t)(row0 + m) * EP + col] = v;
        }
    }
}

// --- serial scan: RK4 neural-ODE + GRU, 16 batch rows per workgroup -------
__global__ __launch_bounds__(256) void recur_kernel(
    const float* __restrict__ g_all, const bf16* __restrict__ W1p,
    const bf16* __restrict__ W2p, const bf16* __restrict__ W3p,
    const bf16* __restrict__ Wup, const float* __restrict__ b_u,
    const float* __restrict__ W_ih, const float* __restrict__ W_hh,
    const float* __restrict__ b_ih, const float* __restrict__ b_hh,
    float* __restrict__ hemb_all) {
    __shared__ float h[16][HP];
    __shared__ float ksum[16][HP];
    __shared__ float kbuf[16][HP];
    __shared__ bf16 abuf[16][HP];
    __shared__ bf16 bbuf[16][HP];
    __shared__ float gx[16][96];
    __shared__ float gh[16][96];
    const int NELEM = 16 * HP;
    int row0 = blockIdx.x * 16;
    const float dt = DTc;

    for (int i = threadIdx.x; i < NELEM; i += 256) {
        int m = i / HP, c = i % HP;
        float v = 0.0f;
        if (c >= Mdim && c < Hdim)
            v = g_all[((size_t)(row0 + m) * Tlen + 0) * EP + (c - Mdim)];
        h[m][c] = v;
    }
    __syncthreads();

    for (int k = 1; k < Tlen; ++k) {
        for (int s = 0; s < NSTEP; ++s) {
            // ---- k1
            for (int i = threadIdx.x; i < NELEM; i += 256) {
                int m = i / HP, c = i % HP;
                abuf[m][c] = (bf16)h[m][c];
            }
            __syncthreads();
            gemm_tile(&abuf[0][0], HP, W1p, 11, 22, &bbuf[0][0], nullptr, 0, 0);
            __syncthreads();
            gemm_tile(&bbuf[0][0], HP, W2p, 11, 22, &abuf[0][0], nullptr, 0, 0);
            __syncthreads();
            gemm_tile(&abuf[0][0], HP, W3p, 11, 22, nullptr, &kbuf[0][0], HP, 1);
            __syncthreads();
            for (int i = threadIdx.x; i < NELEM; i += 256) {
                int m = i / HP, c = i % HP;
                float kk = kbuf[m][c];
                ksum[m][c] = kk;
                abuf[m][c] = (bf16)(h[m][c] + 0.5f * dt * kk);
            }
            __syncthreads();
            // ---- k2
            gemm_tile(&abuf[0][0], HP, W1p, 11, 22, &bbuf[0][0], nullptr, 0, 0);
            __syncthreads();
            gemm_tile(&bbuf[0][0], HP, W2p, 11, 22, &abuf[0][0], nullptr, 0, 0);
            __syncthreads();
            gemm_tile(&abuf[0][0], HP, W3p, 11, 22, nullptr, &kbuf[0][0], HP, 1);
            __syncthreads();
            for (int i = threadIdx.x; i < NELEM; i += 256) {
                int m = i / HP, c = i % HP;
                float kk = kbuf[m][c];
                ksum[m][c] += 2.0f * kk;
                abuf[m][c] = (bf16)(h[m][c] + 0.5f * dt * kk);
            }
            __syncthreads();
            // ---- k3
            gemm_tile(&abuf[0][0], HP, W1p, 11, 22, &bbuf[0][0], nullptr, 0, 0);
            __syncthreads();
            gemm_tile(&bbuf[0][0], HP, W2p, 11, 22, &abuf[0][0], nullptr, 0, 0);
            __syncthreads();
            gemm_tile(&abuf[0][0], HP, W3p, 11, 22, nullptr, &kbuf[0][0], HP, 1);
            __syncthreads();
            for (int i = threadIdx.x; i < NELEM; i += 256) {
                int m = i / HP, c = i % HP;
                float kk = kbuf[m][c];
                ksum[m][c] += 2.0f * kk;
                abuf[m][c] = (bf16)(h[m][c] + dt * kk);
            }
            __syncthreads();
            // ---- k4 + combine
            gemm_tile(&abuf[0][0], HP, W1p, 11, 22, &bbuf[0][0], nullptr, 0, 0);
            __syncthreads();
            gemm_tile(&bbuf[0][0], HP, W2p, 11, 22, &abuf[0][0], nullptr, 0, 0);
            __syncthreads();
            gemm_tile(&abuf[0][0], HP, W3p, 11, 22, nullptr, &kbuf[0][0], HP, 1);
            __syncthreads();
            for (int i = threadIdx.x; i < NELEM; i += 256) {
                int m = i / HP, c = i % HP;
                h[m][c] += (dt / 6.0f) * (ksum[m][c] + kbuf[m][c]);
            }
            __syncthreads();
        }
        // dump h_emb for the decoder pass
        for (int i = threadIdx.x; i < 16 * EP; i += 256) {
            int m = i / EP, c = i % EP;
            float v = (c < Edim) ? h[m][Mdim + c] : 0.0f;
            hemb_all[((size_t)(row0 + m) * (Tlen - 1) + (k - 1)) * EP + c] = v;
        }
        // x = concat(h_mem, g_k) @ W_u^T  (WMMA), gates via VALU
        for (int i = threadIdx.x; i < NELEM; i += 256) {
            int m = i / HP, c = i % HP;
            float v = 0.0f;
            if (c < Mdim)
                v = h[m][c];
            else if (c < Hdim)
                v = g_all[((size_t)(row0 + m) * Tlen + k) * EP + (c - Mdim)];
            abuf[m][c] = (bf16)v;
        }
        __syncthreads();
        gemm_tile(&abuf[0][0], HP, Wup, 11, 20, nullptr, &kbuf[0][0], HP, 2);
        __syncthreads();
        for (int i = threadIdx.x; i < 16 * 90; i += 256) {
            int m = i / 90, j = i % 90;
            float sx = b_ih[j];
            for (int c = 0; c < Edim; ++c)
                sx += (kbuf[m][c] + b_u[c]) * W_ih[j * Edim + c];
            gx[m][j] = sx;
            float sh = b_hh[j];
            for (int c = 0; c < Mdim; ++c) sh += h[m][c] * W_hh[j * Mdim + c];
            gh[m][j] = sh;
        }
        __syncthreads();
        for (int i = threadIdx.x; i < 16 * Mdim; i += 256) {
            int m = i / Mdim, j = i % Mdim;
            float r = 1.0f / (1.0f + __expf(-(gx[m][j] + gh[m][j])));
            float z = 1.0f / (1.0f + __expf(-(gx[m][30 + j] + gh[m][30 + j])));
            float n = tanhf(gx[m][60 + j] + r * gh[m][60 + j]);
            h[m][j] = (1.0f - z) * n + z * h[m][j];
        }
        __syncthreads();
    }
}

// --- decoder: logits = leaky_relu(hemb@Wd1^T+b1)@Wd2^T+b2, masked ---------
__global__ __launch_bounds__(256) void dec_kernel(
    const float* __restrict__ hemb_all, const bf16* __restrict__ Wd1p,
    const float* __restrict__ b_d1, const bf16* __restrict__ Wd2p,
    const float* __restrict__ b_d2, const int* __restrict__ lengths,
    float* __restrict__ out) {
    __shared__ bf16 a0[16][EP];
    __shared__ bf16 h1[16][EP];
    int row0 = blockIdx.x * 16;
    int wave = wave_id();                         // SGPR
    int lane = threadIdx.x & 31;
    for (int i = threadIdx.x; i < 16 * EP; i += 256) {
        int m = i / EP, c = i % EP;
        a0[m][c] = (bf16)hemb_all[(size_t)(row0 + m) * EP + c];
    }
    __syncthreads();
    for (int nt = wave; nt < 20; nt += 8) {
        v8f acc = {};
        for (int kc = 0; kc < 10; ++kc) {
            v16bf a = load_a_frag(&a0[0][0], EP, kc * 32);
            v16bf b = load_b_frag(Wd1p + (((size_t)(kc * 20 + nt)) << 9));
            acc = wmma_bf16(a, b, acc);
        }
        int col = nt * 16 + (lane & 15);
        float bias = (col < Edim) ? b_d1[col] : 0.0f;
#pragma unroll
        for (int r = 0; r < 8; ++r) {
            int m = r + 8 * (lane >> 4);
            float v = acc[r] + bias;
            v = (v > 0.0f) ? v : 0.01f * v;       // leaky_relu
            if (col >= Edim) v = 0.0f;
            h1[m][col] = (bf16)v;
        }
    }
    __syncthreads();
    for (int nt = wave; nt < Vocab / 16; nt += 8) {
        v8f acc = {};
        for (int kc = 0; kc < 10; ++kc) {
            v16bf a = load_a_frag(&h1[0][0], EP, kc * 32);
            v16bf b = load_b_frag(Wd2p + (((size_t)(kc * (Vocab / 16) + nt)) << 9));
            acc = wmma_bf16(a, b, acc);
        }
        int col = nt * 16 + (lane & 15);
        float bias = b_d2[col];
#pragma unroll
        for (int r = 0; r < 8; ++r) {
            int m = r + 8 * (lane >> 4);
            int row = row0 + m;
            int b_ = row / (Tlen - 1), t = row % (Tlen - 1);
            bool valid = (t + 1) < lengths[b_];
            out[(size_t)row * Vocab + col] = valid ? (acc[r] + bias) : 0.0f;
        }
    }
}

extern "C" void kernel_launch(void* const* d_in, const int* in_sizes, int n_in,
                              void* d_out, int out_size, void* d_ws, size_t ws_size,
                              hipStream_t stream) {
    (void)in_sizes; (void)n_in; (void)out_size; (void)ws_size;
    const float* codes  = (const float*)d_in[1];
    const int*   lengths= (const int*)d_in[2];
    const float* W_emb  = (const float*)d_in[3];
    const float* b_emb  = (const float*)d_in[4];
    const float* W_ode1 = (const float*)d_in[5];
    const float* W_ode2 = (const float*)d_in[6];
    const float* W_ode3 = (const float*)d_in[7];
    const float* W_u    = (const float*)d_in[8];
    const float* b_u    = (const float*)d_in[9];
    const float* W_ih   = (const float*)d_in[10];
    const float* W_hh   = (const float*)d_in[11];
    const float* b_ih   = (const float*)d_in[12];
    const float* b_hh   = (const float*)d_in[13];
    const float* W_d1   = (const float*)d_in[14];
    const float* b_d1   = (const float*)d_in[15];
    const float* W_d2   = (const float*)d_in[16];
    const float* b_d2   = (const float*)d_in[17];
    float* out = (float*)d_out;

    char* ws = (char*)d_ws;
    size_t off = 0;
    auto alloc = [&](size_t bytes) -> void* {
        void* p = ws + off;
        off += (bytes + 255) & ~(size_t)255;
        return p;
    };
    float* g_all = (float*)alloc((size_t)Bsz * Tlen * EP * 4);
    float* hemb  = (float*)alloc((size_t)Bsz * (Tlen - 1) * EP * 4);
    bf16* W1p   = (bf16*)alloc((size_t)11 * 22 * 512 * 2);
    bf16* W2p   = (bf16*)alloc((size_t)11 * 22 * 512 * 2);
    bf16* W3p   = (bf16*)alloc((size_t)11 * 22 * 512 * 2);
    bf16* Wembp = (bf16*)alloc((size_t)(Vocab / 32) * 20 * 512 * 2);
    bf16* Wup   = (bf16*)alloc((size_t)11 * 20 * 512 * 2);
    bf16* Wd1p  = (bf16*)alloc((size_t)10 * 20 * 512 * 2);
    bf16* Wd2p  = (bf16*)alloc((size_t)10 * (Vocab / 16) * 512 * 2);

    auto packGrid = [](int ktiles, int ntiles) {
        int total = ktiles * ntiles * 512;
        return (total + 255) / 256;
    };
    pack_b<<<packGrid(11, 22), 256, 0, stream>>>(W_ode1, W1p, Hdim, Hdim, 11, 22);
    pack_b<<<packGrid(11, 22), 256, 0, stream>>>(W_ode2, W2p, Hdim, Hdim, 11, 22);
    pack_b<<<packGrid(11, 22), 256, 0, stream>>>(W_ode3, W3p, Hdim, Hdim, 11, 22);
    pack_b<<<packGrid(Vocab / 32, 20), 256, 0, stream>>>(W_emb, Wembp, Vocab, Edim,
                                                         Vocab / 32, 20);
    pack_b<<<packGrid(11, 20), 256, 0, stream>>>(W_u, Wup, Hdim, Edim, 11, 20);
    pack_b<<<packGrid(10, 20), 256, 0, stream>>>(W_d1, Wd1p, Edim, Edim, 10, 20);
    pack_b<<<packGrid(10, Vocab / 16), 256, 0, stream>>>(W_d2, Wd2p, Edim, Vocab, 10,
                                                         Vocab / 16);

    embed_kernel<<<(Bsz * Tlen) / 16, 256, 0, stream>>>(codes, Wembp, b_emb, g_all);
    recur_kernel<<<Bsz / 16, 256, 0, stream>>>(g_all, W1p, W2p, W3p, Wup, b_u, W_ih,
                                               W_hh, b_ih, b_hh, hemb);
    dec_kernel<<<(Bsz * (Tlen - 1)) / 16, 256, 0, stream>>>(hemb, Wd1p, b_d1, Wd2p,
                                                            b_d2, lengths, out);
}
